// Qwen3Attention_32006096290265
// MI455X (gfx1250) — matrix operand order repaired and checked by reading, not compiled, not run on hardware
//
#include <hip/hip_runtime.h>
#include <hip/hip_bf16.h>

// ---------------------------------------------------------------- types
typedef __bf16 bf16_t;
typedef bf16_t v16bf __attribute__((ext_vector_type(16)));
typedef bf16_t v8bf  __attribute__((ext_vector_type(8)));
typedef bf16_t v4bf  __attribute__((ext_vector_type(4)));
typedef float  v8f   __attribute__((ext_vector_type(8)));

#define T_SEQ 2048
#define HID   2048
#define NH    16
#define NKV   8
#define DH    128
#define NQKV  4096   // NH*DH + 2*NKV*DH

#define WMMA_BF16(a, b, c) \
  __builtin_amdgcn_wmma_f32_16x16x32_bf16(false, (a), false, (b), (short)0, (c), false, false)

// lane-XOR broadcast via LDS crossbar (no LDS memory touched)
__device__ __forceinline__ float xor_f(float v, int m) {
  const int lane = (int)__builtin_amdgcn_mbcnt_lo(~0u, 0u);
  return __int_as_float(
      __builtin_amdgcn_ds_bpermute(((lane ^ m) << 2), __float_as_int(v)));
}

// A-operand fragment, 16x32 bf16 (M x K), memory row-major with K contiguous.
// lane L: row = L%16 ; K chunks [kb, kb+7] and [kb+16, kb+23], kb = (L/16)*8
__device__ __forceinline__ v16bf load_a(const bf16_t* __restrict__ base, int ld, int lane) {
  const int row = lane & 15;
  const int kb  = (lane >> 4) << 3;
  const bf16_t* p = base + (size_t)row * ld + kb;
  v8bf lo = *(const v8bf*)(p);
  v8bf hi = *(const v8bf*)(p + 16);
  v16bf r;
#pragma unroll
  for (int i = 0; i < 8; ++i) { r[i] = lo[i]; r[i + 8] = hi[i]; }
  return r;
}

// B-operand fragment, 32x16 bf16 (K x N), memory: N rows each K-contiguous (NT gemm).
// lane L: col = L%16 ; K chunks [kb, kb+7] and [kb+16, kb+23], kb = (L/16)*8
__device__ __forceinline__ v16bf load_b(const bf16_t* __restrict__ base, int ld, int lane) {
  const int col = lane & 15;
  const int kb  = (lane >> 4) << 3;
  const bf16_t* p = base + (size_t)col * ld + kb;
  v8bf lo = *(const v8bf*)(p);
  v8bf hi = *(const v8bf*)(p + 16);
  v16bf r;
#pragma unroll
  for (int i = 0; i < 8; ++i) { r[i] = lo[i]; r[i + 8] = hi[i]; }
  return r;
}

// ---------------------------------------------------------------- convert
__global__ void to_bf16_kernel(const float* __restrict__ src,
                               bf16_t* __restrict__ dst, int n4) {
  int i = blockIdx.x * blockDim.x + threadIdx.x;
  if (i < n4) {
    const float4 v = ((const float4*)src)[i];
    v4bf o;
    o[0] = (bf16_t)v.x; o[1] = (bf16_t)v.y; o[2] = (bf16_t)v.z; o[3] = (bf16_t)v.w;
    ((v4bf*)dst)[i] = o;
  }
}

// ---------------------------------------------------------------- GEMM (NT)
// C[M,N] = A[M,K] * B[N,K]^T ; 128x128 block tile, 4 waves of 64x64.
__global__ __launch_bounds__(128) void gemm_bf16_nt(
    const bf16_t* __restrict__ A, const bf16_t* __restrict__ B,
    float* __restrict__ C, int K, int ldA, int ldB, int ldC) {
  const int lane = threadIdx.x & 31;
  const int w    = threadIdx.x >> 5;
  const int m0   = blockIdx.y * 128 + (w >> 1) * 64;
  const int n0   = blockIdx.x * 128 + (w & 1) * 64;

  v8f c[4][4];
#pragma unroll
  for (int i = 0; i < 4; ++i)
#pragma unroll
    for (int j = 0; j < 4; ++j)
#pragma unroll
      for (int e = 0; e < 8; ++e) c[i][j][e] = 0.0f;

  const bf16_t* Abase = A + (size_t)m0 * ldA;
  const bf16_t* Bbase = B + (size_t)n0 * ldB;

  for (int k0 = 0; k0 < K; k0 += 32) {
    if (k0 + 32 < K) {   // next-slice prefetch -> global_prefetch_b8
      __builtin_prefetch(Abase + (size_t)(lane & 15) * ldA + k0 + 32, 0, 1);
      __builtin_prefetch(Bbase + (size_t)(lane & 15) * ldB + k0 + 32, 0, 1);
    }
    v16bf a[4], b[4];
#pragma unroll
    for (int i = 0; i < 4; ++i)
      a[i] = load_a(Abase + (size_t)(16 * i) * ldA + k0, ldA, lane);
#pragma unroll
    for (int j = 0; j < 4; ++j)
      b[j] = load_b(Bbase + (size_t)(16 * j) * ldB + k0, ldB, lane);
#pragma unroll
    for (int i = 0; i < 4; ++i)
#pragma unroll
      for (int j = 0; j < 4; ++j)
        c[i][j] = WMMA_BF16(a[i], b[j], c[i][j]);
  }

  const int half = lane >> 4, col = lane & 15;
#pragma unroll
  for (int i = 0; i < 4; ++i)
#pragma unroll
    for (int j = 0; j < 4; ++j) {
      float* cp = C + (size_t)(m0 + 16 * i + 8 * half) * ldC + (n0 + 16 * j + col);
#pragma unroll
      for (int r = 0; r < 8; ++r) cp[(size_t)r * ldC] = c[i][j][r];
    }
}

// ---------------------------------------------------------------- norm+rope
// grid (T, NH + 2*NKV), 32 threads. y<16: q ; 16..23: k ; 24..31: v (cast+transpose)
__global__ __launch_bounds__(32) void norm_rope_kernel(
    const float* __restrict__ qkv, const int* __restrict__ pos,
    const float* __restrict__ qw, const float* __restrict__ kw,
    bf16_t* __restrict__ qb, bf16_t* __restrict__ kb, bf16_t* __restrict__ vT) {
  const int t = blockIdx.x;
  const int h = blockIdx.y;
  const int lane = threadIdx.x;

  if (h >= NH + NKV) {                       // V: cast + transpose
    const int kvh = h - NH - NKV;
    const float* src = qkv + (size_t)t * NQKV + NH * DH + NKV * DH + kvh * DH;
#pragma unroll
    for (int i = 0; i < 4; ++i) {
      const int d = lane + 32 * i;
      vT[((size_t)kvh * DH + d) * T_SEQ + t] = (bf16_t)src[d];
    }
    return;
  }

  const bool isq = (h < NH);
  const float* src = qkv + (size_t)t * NQKV + (isq ? h * DH : NH * DH + (h - NH) * DH);
  float x[4];
#pragma unroll
  for (int i = 0; i < 4; ++i) x[i] = src[lane + 32 * i];

  float ss = x[0] * x[0] + x[1] * x[1] + x[2] * x[2] + x[3] * x[3];
#pragma unroll
  for (int mshift = 0; mshift < 5; ++mshift) ss += xor_f(ss, 1 << mshift);
  const float rms = rsqrtf(ss * (1.0f / DH) + 1e-6f);

  const float* wv = isq ? qw : kw;
  float xn[4];
#pragma unroll
  for (int i = 0; i < 4; ++i) xn[i] = x[i] * rms * wv[lane + 32 * i];

  const float p = (float)pos[t];
  // pair a: j = lane   -> (xn[0], xn[2]) ; pair b: j = lane+32 -> (xn[1], xn[3])
  const float L2T = 19.931568569324174f;     // log2(1e6)
  const float fa = exp2f(-(float)lane * (L2T / 64.0f));
  const float fb = exp2f(-(float)(lane + 32) * (L2T / 64.0f));
  float sa, ca, sb, cb;
  __sincosf(p * fa, &sa, &ca);
  __sincosf(p * fb, &sb, &cb);
  const float oa1 = xn[0] * ca - xn[2] * sa;
  const float oa2 = xn[2] * ca + xn[0] * sa;
  const float ob1 = xn[1] * cb - xn[3] * sb;
  const float ob2 = xn[3] * cb + xn[1] * sb;

  bf16_t* dst = isq ? (qb + ((size_t)h * T_SEQ + t) * DH)
                    : (kb + ((size_t)(h - NH) * T_SEQ + t) * DH);
  dst[lane]      = (bf16_t)oa1;
  dst[lane + 64] = (bf16_t)oa2;
  dst[lane + 32] = (bf16_t)ob1;
  dst[lane + 96] = (bf16_t)ob2;
}

// ---------------------------------------------------------------- attention
// 1 wave = (head, 16 query rows). Flash online-softmax, 32 keys / iteration.
__global__ __launch_bounds__(128) void attn_kernel(
    const bf16_t* __restrict__ qb, const bf16_t* __restrict__ kb,
    const bf16_t* __restrict__ vT, bf16_t* __restrict__ ob) {
  __shared__ bf16_t plds[4][16 * 32];
  const int lane = threadIdx.x & 31;
  const int w    = threadIdx.x >> 5;
  const int widx = blockIdx.x * 4 + w;
  const int head = widx >> 7;                // widx / (T/16)
  const int q0   = (widx & 127) << 4;
  const int kvh  = head >> 1;                // GQA rep = 2

  const bf16_t* Q  = qb + ((size_t)head * T_SEQ + q0) * DH;
  const bf16_t* Kp = kb + (size_t)kvh * T_SEQ * DH;
  const bf16_t* Vp = vT + (size_t)kvh * DH * T_SEQ;

  v16bf qf[4];
#pragma unroll
  for (int f = 0; f < 4; ++f) qf[f] = load_a(Q + 32 * f, DH, lane);

  v8f o[8];
#pragma unroll
  for (int dt = 0; dt < 8; ++dt)
#pragma unroll
    for (int e = 0; e < 8; ++e) o[dt][e] = 0.0f;
  float m[8], l[8];
#pragma unroll
  for (int r = 0; r < 8; ++r) { m[r] = -3.0e38f; l[r] = 0.0f; }

  const int half = lane >> 4, coln = lane & 15;
  const float scale = 0.088388347648318447f;   // 1/sqrt(128)

  for (int s0base = 0; s0base < q0 + 16; s0base += 32) {
    v8f s0, s1;
#pragma unroll
    for (int e = 0; e < 8; ++e) { s0[e] = 0.0f; s1[e] = 0.0f; }
#pragma unroll
    for (int f = 0; f < 4; ++f) {
      v16bf kf = load_b(Kp + (size_t)s0base * DH + 32 * f, DH, lane);
      s0 = WMMA_BF16(qf[f], kf, s0);
    }
#pragma unroll
    for (int f = 0; f < 4; ++f) {
      v16bf kf = load_b(Kp + (size_t)(s0base + 16) * DH + 32 * f, DH, lane);
      s1 = WMMA_BF16(qf[f], kf, s1);
    }

    float corr[8];
#pragma unroll
    for (int r = 0; r < 8; ++r) {
      const int qrow = q0 + r + 8 * half;
      float v0 = s0[r] * scale;
      float v1 = s1[r] * scale;
      if (s0base + coln > qrow)      v0 = -1.0e30f;
      if (s0base + 16 + coln > qrow) v1 = -1.0e30f;
      float mx = fmaxf(v0, v1);
#pragma unroll
      for (int d = 1; d < 16; d <<= 1) mx = fmaxf(mx, xor_f(mx, d));
      const float mn = fmaxf(m[r], mx);
      const float cr = __expf(m[r] - mn);
      const float p0 = __expf(v0 - mn);
      const float p1 = __expf(v1 - mn);
      float sum = p0 + p1;
#pragma unroll
      for (int d = 1; d < 16; d <<= 1) sum += xor_f(sum, d);
      l[r] = l[r] * cr + sum;
      m[r] = mn;
      corr[r] = cr;
      const int prow = r + 8 * half;
      plds[w][prow * 32 + coln]      = (bf16_t)p0;
      plds[w][prow * 32 + 16 + coln] = (bf16_t)p1;
    }
#pragma unroll
    for (int dt = 0; dt < 8; ++dt)
#pragma unroll
      for (int r = 0; r < 8; ++r) o[dt][r] *= corr[r];

    asm volatile("s_wait_dscnt 0" ::: "memory");  // DS in-order per wave; be explicit

    // reload P (16x32) from LDS in A-fragment layout
    v16bf pf;
    {
      const int prow = lane & 15;
      const int kb2  = (lane >> 4) << 3;
      const bf16_t* pp = &plds[w][prow * 32 + kb2];
      v8bf lo = *(const v8bf*)(pp);
      v8bf hi = *(const v8bf*)(pp + 16);
#pragma unroll
      for (int i = 0; i < 8; ++i) { pf[i] = lo[i]; pf[i + 8] = hi[i]; }
    }
#pragma unroll
    for (int dt = 0; dt < 8; ++dt) {
      v16bf vf = load_b(Vp + (size_t)(dt * 16) * T_SEQ + s0base, T_SEQ, lane);
      o[dt] = WMMA_BF16(pf, vf, o[dt]);
    }
  }

  float inv[8];
#pragma unroll
  for (int r = 0; r < 8; ++r) inv[r] = 1.0f / l[r];
#pragma unroll
  for (int dt = 0; dt < 8; ++dt)
#pragma unroll
    for (int r = 0; r < 8; ++r)
      ob[(size_t)(q0 + r + 8 * half) * (NH * DH) + head * DH + dt * 16 + coln] =
          (bf16_t)(o[dt][r] * inv[r]);
}

// ---------------------------------------------------------------- launch
extern "C" void kernel_launch(void* const* d_in, const int* in_sizes, int n_in,
                              void* d_out, int out_size, void* d_ws, size_t ws_size,
                              hipStream_t stream) {
  (void)in_sizes; (void)n_in; (void)out_size; (void)ws_size;
  const float* hidden = (const float*)d_in[0];
  const int*   pos    = (const int*)d_in[1];
  const float* qkv_w  = (const float*)d_in[2];
  const float* o_w    = (const float*)d_in[3];
  const float* qnw    = (const float*)d_in[4];
  const float* knw    = (const float*)d_in[5];
  float* out = (float*)d_out;

  char* ws = (char*)d_ws;
  bf16_t* Xb  = (bf16_t*)(ws);                       //  8 MB  hidden bf16
  bf16_t* Wb  = (bf16_t*)(ws + (8u  << 20));         // 16 MB  qkv_w bf16
  bf16_t* Owb = (bf16_t*)(ws + (24u << 20));         //  8 MB  o_w bf16
  float*  qkv = (float*) (ws + (32u << 20));         // 32 MB  qkv f32
  bf16_t* qb  = (bf16_t*)(ws + (64u << 20));         //  8 MB  q [H,T,D]
  bf16_t* kbf = (bf16_t*)(ws + (72u << 20));         //  4 MB  k [KV,T,D]
  bf16_t* vT  = (bf16_t*)(ws + (76u << 20));         //  4 MB  v^T [KV,D,T]
  bf16_t* ob  = (bf16_t*)(ws + (80u << 20));         //  8 MB  attn out bf16

  // 1) converts
  {
    int n4 = (T_SEQ * HID) / 4;
    to_bf16_kernel<<<(n4 + 255) / 256, 256, 0, stream>>>(hidden, Xb, n4);
    n4 = (NQKV * HID) / 4;
    to_bf16_kernel<<<(n4 + 255) / 256, 256, 0, stream>>>(qkv_w, Wb, n4);
    n4 = (HID * NH * DH) / 4;
    to_bf16_kernel<<<(n4 + 255) / 256, 256, 0, stream>>>(o_w, Owb, n4);
  }
  // 2) qkv = hidden @ qkv_w^T   [2048 x 4096]
  gemm_bf16_nt<<<dim3(NQKV / 128, T_SEQ / 128), 128, 0, stream>>>(
      Xb, Wb, qkv, HID, HID, HID, NQKV);
  // 3) rmsnorm + rope + relayout
  norm_rope_kernel<<<dim3(T_SEQ, NH + 2 * NKV), 32, 0, stream>>>(
      qkv, pos, qnw, knw, qb, kbf, vT);
  // 4) causal GQA flash attention
  attn_kernel<<<(NH * (T_SEQ / 16)) / 4, 128, 0, stream>>>(qb, kbf, vT, ob);
  // 5) out = attn_out @ o_w^T   [2048 x 2048]
  gemm_bf16_nt<<<dim3(HID / 128, T_SEQ / 128), 128, 0, stream>>>(
      ob, Owb, out, NH * DH, NH * DH, NH * DH, HID);
}
